// SMultiHeadAttention_386547057194
// MI455X (gfx1250) — matrix operand order repaired and checked by reading, not compiled, not run on hardware
//
#include <hip/hip_runtime.h>
#include <hip/hip_bf16.h>

typedef __attribute__((ext_vector_type(16))) _Float16 v16h;
typedef __attribute__((ext_vector_type(8)))  _Float16 h8;
typedef __attribute__((ext_vector_type(8)))  float    v8f;

#define WMMA_F32_F16(a, b, c) \
    __builtin_amdgcn_wmma_f32_16x16x32_f16(false, (a), false, (b), (short)0, (c), false, false)

constexpr int Bb = 4, Nn = 512, Tt = 24, Cc = 64, Hh = 4, Dd = 16;
constexpr int ROWS     = Bb * Nn * Tt;   // 49152 rows of C=64
constexpr int ROWTILES = ROWS / 16;      // 3072
constexpr int NTILES   = Nn / 16;        // 32
constexpr int PROB     = Bb * Hh * Tt;   // 384 attention problems
constexpr int SCPAD    = Nn + 4;         // f32 row stride: 516 ≡ 4 mod 64 banks
constexpr int ATPAD    = Nn + 8;         // f16 row stride: 520 (16B-aligned rows)

union U16h { v16h v; h8 h[2]; };

// ---------------------------------------------------------------------------
// Projection: y[r,o] = sum_c x[r,c] * W[o,c] -> f16.
// TRANSPOSED=false : store [B,H,T,N,D]  (Q, K)
// TRANSPOSED=true  : store [B,H,T,D,N]  (V, k-major for the P@V B-fragment;
//                    also turns the store into one packed 16B store per lane)
// ---------------------------------------------------------------------------
template <bool TRANSPOSED>
__global__ void proj_qkv_kernel(const float* __restrict__ X,
                                const float* __restrict__ W,
                                _Float16* __restrict__ outw)
{
    const int lane = threadIdx.x;
    const int l16  = lane & 15;
    const int hi   = lane >> 4;               // 0 or 1 (lane half)
    const int nt   = blockIdx.x & (NTILES-1); // n tile 0..31
    const int t    = (blockIdx.x >> 5) % Tt;  // 0..23   (scalar div, once)
    const int bb   = blockIdx.x / (NTILES*Tt);
    const int head = blockIdx.y;              // 16-col tile == one head

    // A rows: x[b, nt*16 + row, t, :]  (row = l16)
    const float* xrow = X + (((size_t)bb * Nn + nt*16 + l16) * Tt + t) * Cc;
    const float* wrow = W + (size_t)(head * 16 + l16) * Cc;

    v8f acc = {};
    #pragma unroll
    for (int cc = 0; cc < 2; ++cc) {
        v16h a, b;
        // A: halves 0..15 <-> k = hi*16 + 0..15
        #pragma unroll
        for (int g = 0; g < 2; ++g) {
            float4 f0 = *(const float4*)(xrow + cc * 32 + hi * 16 + g * 8);
            float4 f1 = *(const float4*)(xrow + cc * 32 + hi * 16 + g * 8 + 4);
            a[g*8 + 0] = (_Float16)f0.x; a[g*8 + 1] = (_Float16)f0.y;
            a[g*8 + 2] = (_Float16)f0.z; a[g*8 + 3] = (_Float16)f0.w;
            a[g*8 + 4] = (_Float16)f1.x; a[g*8 + 5] = (_Float16)f1.y;
            a[g*8 + 6] = (_Float16)f1.z; a[g*8 + 7] = (_Float16)f1.w;
        }
        // B: col o = head*16 + l16, k = h + hi*16 (16 consecutive c)
        #pragma unroll
        for (int h = 0; h < 16; h += 4) {
            float4 f = *(const float4*)(wrow + cc * 32 + hi * 16 + h);
            b[h]     = (_Float16)f.x;
            b[h + 1] = (_Float16)f.y;
            b[h + 2] = (_Float16)f.z;
            b[h + 3] = (_Float16)f.w;
        }
        acc = WMMA_F32_F16(a, b, acc);
    }

    if (TRANSPOSED) {
        // [B,H,T,D,N]: d = l16, n = nt*16 + r + hi*8 (r consecutive) -> one b128
        union { _Float16 h[8]; uint4 u; } pk;
        #pragma unroll
        for (int r = 0; r < 8; ++r) pk.h[r] = (_Float16)acc[r];
        _Float16* obase = outw +
            ((((size_t)bb * Hh + head) * Tt + t) * Dd + l16) * Nn + nt*16 + hi*8;
        *(uint4*)obase = pk.u;
    } else {
        // [B,H,T,N,D]: n = nt*16 + (r + hi*8), d = l16
        _Float16* obase = outw +
            ((((size_t)bb * Hh + head) * Tt + t) * Nn + nt*16) * Dd;
        #pragma unroll
        for (int r = 0; r < 8; ++r)
            obase[(r + hi * 8) * Dd + l16] = (_Float16)acc[r];
    }
}

// ---------------------------------------------------------------------------
// Fused attention: one wave per (16-query tile, problem p=(b,h,t)).
// Full 512-wide score rows in LDS; 2-pass softmax with 1/sum folded into the
// ctx epilogue via a 16-entry LDS table. V is consumed k-major ([...,D,N]).
// ---------------------------------------------------------------------------
__global__ void attn_kernel(const _Float16* __restrict__ qw,
                            const _Float16* __restrict__ kw,
                            const _Float16* __restrict__ vwT,
                            const float* __restrict__ bias,
                            _Float16* __restrict__ ctxw)
{
    __shared__ float    sc[16][SCPAD];   // scores (f32)
    __shared__ _Float16 at[16][ATPAD];   // unnormalized exp probs (f16)
    __shared__ float    inv_s[16];       // 1/rowsum

    const int lane = threadIdx.x;
    const int l16  = lane & 15;
    const int hi   = lane >> 4;
    const int qt   = blockIdx.x;         // 0..31 query tile
    const int p    = blockIdx.y;         // 0..383 problem
    const int t    = p % Tt;
    const int hd   = (p / Tt) % Hh;
    const int bb   = p / (Tt * Hh);

    // Q fragment (K = D = 16 real, K 16..31 zero-padded): hi=0 lanes halves 0..15 = d
    v16h aq = {};
    {
        U16h u; u.v = aq;
        if (hi == 0) { u.h[0] = *(const h8*)(qw + (((size_t)p * Nn) + qt * 16 + l16) * Dd); u.h[1] = *(const h8*)(qw + (((size_t)p * Nn) + qt * 16 + l16) * Dd + 8); }
        aq = u.v;
    }

    const size_t bias_base = (((size_t)bb * Hh + hd) * Nn) * (size_t)Nn;

    // scores = (Q K^T) * 0.25 + bias.  Two key tiles per trip: the second
    // back-to-back WMMA (independent) fills the WMMA->VALU hazard slots of the
    // first. bk0/bk1 hoisted so the hi=1 zero halves are loop-invariant.
    v16h bk0 = {}, bk1 = {};
    for (int mt = 0; mt < NTILES; mt += 2) {
        if (hi == 0) {  // K dims 0..15 real; lanes 16..31 keep the zero pad
            const _Float16* kr0 = kw + (((size_t)p * Nn) + mt * 16 + l16) * Dd;
            U16h u0, u1;
            u0.h[0] = *(const h8*)(kr0);
            u0.h[1] = *(const h8*)(kr0 + 8);
            u1.h[0] = *(const h8*)(kr0 + 16 * Dd);
            u1.h[1] = *(const h8*)(kr0 + 16 * Dd + 8);
            bk0 = u0.v;
            bk1 = u1.v;
        }
        v8f z0 = {}, z1 = {};
        v8f s0 = WMMA_F32_F16(aq, bk0, z0);
        v8f s1 = WMMA_F32_F16(aq, bk1, z1);
        #pragma unroll
        for (int r = 0; r < 8; ++r) {
            int lrow = r + hi * 8;
            int n    = qt * 16 + lrow;
            int m    = mt * 16 + l16;
            sc[lrow][m] = s0[r] * 0.25f + bias[bias_base + (size_t)n * Nn + m];
        }
        #pragma unroll
        for (int r = 0; r < 8; ++r) {
            int lrow = r + hi * 8;
            int n    = qt * 16 + lrow;
            int m    = (mt + 1) * 16 + l16;
            sc[lrow][m] = s1[r] * 0.25f + bias[bias_base + (size_t)n * Nn + m];
        }
    }
    __syncthreads();

    // Softmax: 2 lanes per row (row = l16, half = hi). float4 LDS sweeps.
    {
        const int row = l16;
        const float4* prow = (const float4*)&sc[row][hi * (Nn / 2)];
        float mx = -3.402823466e38f;
        #pragma unroll 4
        for (int i = 0; i < Nn / 8; ++i) {
            float4 f = prow[i];
            mx = fmaxf(mx, fmaxf(fmaxf(f.x, f.y), fmaxf(f.z, f.w)));
        }
        mx = fmaxf(mx, __shfl_xor(mx, 16, 32));

        float sum = 0.0f;
        _Float16* arow = &at[row][hi * (Nn / 2)];
        #pragma unroll 2
        for (int i = 0; i < Nn / 8; ++i) {
            float4 f = prow[i];
            float e0 = __expf(f.x - mx), e1 = __expf(f.y - mx);
            float e2 = __expf(f.z - mx), e3 = __expf(f.w - mx);
            sum += (e0 + e1) + (e2 + e3);
            union { _Float16 h[4]; uint2 u; } pk;
            pk.h[0] = (_Float16)e0; pk.h[1] = (_Float16)e1;
            pk.h[2] = (_Float16)e2; pk.h[3] = (_Float16)e3;
            *(uint2*)&arow[i * 4] = pk.u;   // 8B-aligned packed store
        }
        sum += __shfl_xor(sum, 16, 32);
        if (hi == 0) inv_s[row] = 1.0f / sum;
    }
    __syncthreads();

    // ctx = P_unnorm @ V : 16 WMMAs over K = 512, split across two independent
    // accumulator chains (even/odd k-tiles) to halve the serial WMMA depth.
    v8f c0 = {}, c1 = {};
    for (int kt = 0; kt < Nn / 32; kt += 2) {
        U16h ua0, ub0, ua1, ub1;
        // A fragment: halves 0..7 <-> k = kt*32 + hi*8 + 0..7 (16B contiguous),
        //             halves 8..15 <-> k = kt*32 + 16 + hi*8 + 0..7
        ua0.h[0] = *(const h8*)&at[l16][kt * 32 + hi * 8];
        ua0.h[1] = *(const h8*)&at[l16][kt * 32 + 16 + hi * 8];
        ua1.h[0] = *(const h8*)&at[l16][kt * 32 + 32 + hi * 8];
        ua1.h[1] = *(const h8*)&at[l16][kt * 32 + 48 + hi * 8];
        // B fragment: col d = l16, same k order as A
        const _Float16* vb = vwT + (((size_t)p * Dd) + l16) * Nn + kt * 32 + hi * 8;
        ub0.h[0] = *(const h8*)(vb);
        ub0.h[1] = *(const h8*)(vb + 16);
        ub1.h[0] = *(const h8*)(vb + 32);
        ub1.h[1] = *(const h8*)(vb + 48);
        c0 = WMMA_F32_F16(ua0.v, ub0.v, c0);
        c1 = WMMA_F32_F16(ua1.v, ub1.v, c1);
    }

    // Normalize by 1/rowsum and store ctx in [B,N,T,C] (o = hd*16 + d).
    _Float16* obase = ctxw + (((size_t)bb * Nn + qt * 16) * Tt + t) * Cc + hd * 16 + l16;
    #pragma unroll
    for (int r = 0; r < 8; ++r) {
        int lrow = r + hi * 8;
        obase[(size_t)lrow * Tt * Cc] = (_Float16)((c0[r] + c1[r]) * inv_s[lrow]);
    }
}

// ---------------------------------------------------------------------------
// Output projection: out[r,o] = sum_c ctx[r,c] * W_O[o,c] + b_O[o]  (f32 out)
// Rows are flat (b,n,t) — contiguous in both ctx and out, no decode needed.
// ---------------------------------------------------------------------------
__global__ void out_proj_kernel(const _Float16* __restrict__ ctxw,
                                const float* __restrict__ W,
                                const float* __restrict__ bO,
                                float* __restrict__ out)
{
    const int lane = threadIdx.x;
    const int l16  = lane & 15;
    const int hi   = lane >> 4;
    const int rowTile = blockIdx.x;
    const int colT    = blockIdx.y;

    const _Float16* xrow = ctxw + (size_t)(rowTile * 16 + l16) * Cc;
    const float*    wrow = W    + (size_t)(colT * 16 + l16) * Cc;

    v8f acc = {};
    #pragma unroll
    for (int cc = 0; cc < 2; ++cc) {
        // A: halves 0..15 <-> k = hi*16 + 0..15
        U16h ua;
        ua.h[0] = *(const h8*)(xrow + cc * 32 + hi * 16);
        ua.h[1] = *(const h8*)(xrow + cc * 32 + hi * 16 + 8);
        v16h b;
        #pragma unroll
        for (int h = 0; h < 16; h += 4) {
            float4 f = *(const float4*)(wrow + cc * 32 + hi * 16 + h);
            b[h]     = (_Float16)f.x;
            b[h + 1] = (_Float16)f.y;
            b[h + 2] = (_Float16)f.z;
            b[h + 3] = (_Float16)f.w;
        }
        acc = WMMA_F32_F16(ua.v, b, acc);
    }

    float bo = bO[colT * 16 + l16];
    #pragma unroll
    for (int r = 0; r < 8; ++r) {
        size_t R = (size_t)rowTile * 16 + r + hi * 8;
        out[R * Cc + colT * 16 + l16] = acc[r] + bo;
    }
}

// ---------------------------------------------------------------------------
extern "C" void kernel_launch(void* const* d_in, const int* in_sizes, int n_in,
                              void* d_out, int out_size, void* d_ws, size_t ws_size,
                              hipStream_t stream)
{
    (void)in_sizes; (void)n_in; (void)out_size; (void)ws_size;

    const float* inQ  = (const float*)d_in[0];
    const float* inK  = (const float*)d_in[1];
    const float* inV  = (const float*)d_in[2];
    const float* bias = (const float*)d_in[3];
    const float* WQ   = (const float*)d_in[4];
    const float* WK   = (const float*)d_in[5];
    const float* WV   = (const float*)d_in[6];
    const float* WO   = (const float*)d_in[7];
    const float* bO   = (const float*)d_in[8];
    float* out = (float*)d_out;

    const size_t per = (size_t)PROB * Nn * Dd;   // 3,145,728 f16 elems / buffer
    _Float16* qw  = (_Float16*)d_ws;
    _Float16* kw  = qw + per;
    _Float16* vwT = kw + per;                    // V in [B,H,T,D,N] (k-major)
    _Float16* cw  = vwT + per;                   // ctx in [B,N,T,C], f16

    dim3 blk(32);
    dim3 gproj(ROWTILES, Hh);

    proj_qkv_kernel<false><<<gproj, blk, 0, stream>>>(inQ, WQ, qw);
    proj_qkv_kernel<false><<<gproj, blk, 0, stream>>>(inK, WK, kw);
    proj_qkv_kernel<true ><<<gproj, blk, 0, stream>>>(inV, WV, vwT);

    attn_kernel<<<dim3(NTILES, PROB), blk, 0, stream>>>(qw, kw, vwT, bias, cw);

    out_proj_kernel<<<gproj, blk, 0, stream>>>(cw, WO, bO, out);
}
